// LmLSTM_70583492543093
// MI455X (gfx1250) — matrix operand order repaired and checked by reading, not compiled, not run on hardware
//
#include <hip/hip_runtime.h>
#include <hip/hip_bf16.h>

// ---------------------------------------------------------------------------
// LSTM LM forward for MI455X (gfx1250, wave32, WMMA).
// All matmuls via v_wmma_f32_16x16x32_bf16 (bf16 in, f32 acc).
// Big GEMMs: 64x128 block tile, wave computes 32x32 (4 WMMAs/K-step),
// double-buffered LDS with register-staged global prefetch (1 barrier/step).
// ---------------------------------------------------------------------------

typedef __attribute__((ext_vector_type(16))) __bf16         v16bf;
typedef __attribute__((ext_vector_type(8)))  float          v8f;
typedef __attribute__((ext_vector_type(16))) unsigned short v16us;
typedef __attribute__((ext_vector_type(8)))  unsigned short v8us;
typedef __attribute__((ext_vector_type(4)))  unsigned short v4us;
typedef __attribute__((ext_vector_type(4)))  float          v4f;
typedef __attribute__((ext_vector_type(2)))  float          v2f;

static constexpr int B_ = 16, T_ = 256, D_ = 1024, V_ = 32000, L_ = 2;
static constexpr int N4_ = 4 * D_;         // 4096
static constexpr int M_  = B_ * T_;        // 4096

__device__ __forceinline__ unsigned short f2bf(float f) {
  unsigned int u = __builtin_bit_cast(unsigned int, f);
  u += 0x7fffu + ((u >> 16) & 1u);          // round-to-nearest-even
  return (unsigned short)(u >> 16);
}

__device__ __forceinline__ float sig_(float x) {
  return 1.0f / (1.0f + __expf(-x));
}
__device__ __forceinline__ float tanh_(float x) {
  x = fminf(fmaxf(x, -15.0f), 15.0f);
  float e = __expf(2.0f * x);
  return (e - 1.0f) / (e + 1.0f);
}

// A fragment: LDS tile As[rows][40] (bf16 as u16), 32 K per step.
// ISA 16-bit A 16x32 layout: lanes 0-15 elems0-7 K=0..7, elems8-15 K=16..23;
// lanes 16-31: K=8..15 / K=24..31.  M = lane&15 (+mBase).
__device__ __forceinline__ v16bf frag_a(const unsigned short* As, int mBase, int lane) {
  int m  = mBase + (lane & 15);
  int hi = (lane >> 4) & 1;
  const unsigned short* row = As + m * 40;
  v8us p0 = *(const v8us*)(row + hi * 8);
  v8us p1 = *(const v8us*)(row + 16 + hi * 8);
  v16us r;
#pragma unroll
  for (int i = 0; i < 8; ++i) { r[i] = p0[i]; r[i + 8] = p1[i]; }
  return __builtin_bit_cast(v16bf, r);
}

// B fragment from LDS transposed tile Bt[n][40] (n-major, K contiguous).
// B 32x16 layout (matches A): lanes 0-15 elems0-7 K=0..7, elems8-15 K=16..23; lanes 16-31: K=8..15 / K=24..31. N = lane&15.
__device__ __forceinline__ v16bf frag_b(const unsigned short* Bt, int nBase, int lane) {
  int n  = nBase + (lane & 15);
  int hi = (lane >> 4) & 1;
  const unsigned short* row = Bt + n * 40;
  v8us p0 = *(const v8us*)(row + hi * 8);
  v8us p1 = *(const v8us*)(row + 16 + hi * 8);
  v16us r;
#pragma unroll
  for (int i = 0; i < 8; ++i) { r[i] = p0[i]; r[i + 8] = p1[i]; }
  return __builtin_bit_cast(v16bf, r);
}

__device__ __forceinline__ v8f wmma_bf16(v16bf a, v16bf b, v8f c) {
  return __builtin_amdgcn_wmma_f32_16x16x32_bf16(false, a, false, b, (short)0, c,
                                                 false, false);
}

// ---------------------------------------------------------------------------
// f32 -> bf16 conversion (vectorized)
// ---------------------------------------------------------------------------
__global__ __launch_bounds__(256) void cvt_bf16_kernel(const float* __restrict__ src,
                                                       unsigned short* __restrict__ dst,
                                                       int n4) {
  int i = blockIdx.x * 256 + threadIdx.x;
  if (i < n4) {
    v4f v = *(const v4f*)(src + (size_t)i * 4);
    v4us h;
#pragma unroll
    for (int j = 0; j < 4; ++j) h[j] = f2bf(v[j]);
    *(v4us*)(dst + (size_t)i * 4) = h;
  }
}

// ---------------------------------------------------------------------------
// Embedding gather: X[row][0:1024] = E[token[row]][:]
// ---------------------------------------------------------------------------
__global__ __launch_bounds__(256) void embed_kernel(const int* __restrict__ tokens,
                                                    const float* __restrict__ E,
                                                    float* __restrict__ X) {
  int row = blockIdx.x;
  int tok = tokens[row];
  int d   = threadIdx.x * 4;
  *(v4f*)(X + (size_t)row * D_ + d) = *(const v4f*)(E + (size_t)tok * D_ + d);
}

// ---------------------------------------------------------------------------
// Big GEMM: Out[M x N] = Ab(bf16)[M x K] @ Wb(bf16)[K x N] + bias
// 256 threads (8 waves), macro tile 64(M) x 128(N), K-step 32.
// Waves 2x4, each wave 32x32 output (4 WMMAs per K-step).
// Double-buffered LDS; next tile staged through registers while WMMAs run.
// MASK: rows with token==0 replaced by onehot(0) (output projection).
// ---------------------------------------------------------------------------
template <int N, bool MASK>
__global__ __launch_bounds__(256) void big_gemm_kernel(
    const unsigned short* __restrict__ Ab,  // [4096 x 1024] bf16
    const unsigned short* __restrict__ Wb,  // [1024 x N] bf16
    const float* __restrict__ bias,         // [N]
    const int* __restrict__ tokens,         // [4096] (only if MASK)
    float* __restrict__ Out) {              // [4096 x N]
  constexpr int K = D_;
  __shared__ __align__(16) unsigned short As[2][64 * 40];
  __shared__ __align__(16) unsigned short Bt[2][128 * 40];
  const int tid = threadIdx.x, lane = tid & 31, wid = tid >> 5;
  const int m0 = blockIdx.y * 64, n0 = blockIdx.x * 128;
  const int wM = (wid >> 2) * 32;   // 0 or 32
  const int wN = (wid & 3) * 32;    // 0,32,64,96
  const int hi = (lane >> 4) & 1;
  const int l15 = lane & 15;

  // per-thread staging coordinates (fixed across K-steps)
  int aR[2], aK[2], bK[4], bN[4];
#pragma unroll
  for (int it = 0; it < 2; ++it) {
    int idx = (tid + it * 256) * 4;
    aR[it] = idx >> 5; aK[it] = idx & 31;          // 64 x 32 A tile
  }
#pragma unroll
  for (int it = 0; it < 4; ++it) {
    int idx = (tid + it * 256) * 4;
    bK[it] = idx >> 7; bN[it] = idx & 127;         // 32 x 128 B tile
  }

  v8f acc[2][2];
#pragma unroll
  for (int j = 0; j < 2; ++j) {
    float bv = bias[n0 + wN + j * 16 + l15];
#pragma unroll
    for (int i = 0; i < 2; ++i)
#pragma unroll
      for (int r = 0; r < 8; ++r) acc[i][j][r] = bv;
  }

  v4us aReg[2], bReg[4];
  // prologue: load + store tile 0
#pragma unroll
  for (int it = 0; it < 2; ++it)
    aReg[it] = *(const v4us*)(Ab + (size_t)(m0 + aR[it]) * K + aK[it]);
#pragma unroll
  for (int it = 0; it < 4; ++it)
    bReg[it] = *(const v4us*)(Wb + (size_t)bK[it] * N + n0 + bN[it]);
#pragma unroll
  for (int it = 0; it < 2; ++it)
    *(v4us*)(&As[0][aR[it] * 40 + aK[it]]) = aReg[it];
#pragma unroll
  for (int it = 0; it < 4; ++it)
#pragma unroll
    for (int i = 0; i < 4; ++i) Bt[0][(bN[it] + i) * 40 + bK[it]] = bReg[it][i];
  __syncthreads();

  for (int k0 = 0; k0 < K; k0 += 32) {
    const int cur = (k0 >> 5) & 1;
    const bool more = (k0 + 32) < K;
    if (more) {  // issue global loads for next tile (overlap with WMMA)
      int kk = k0 + 32;
#pragma unroll
      for (int it = 0; it < 2; ++it)
        aReg[it] = *(const v4us*)(Ab + (size_t)(m0 + aR[it]) * K + kk + aK[it]);
#pragma unroll
      for (int it = 0; it < 4; ++it)
        bReg[it] = *(const v4us*)(Wb + (size_t)(kk + bK[it]) * N + n0 + bN[it]);
      if (kk + 32 < K)
        __builtin_prefetch((const void*)(Wb + (size_t)(kk + 32) * N + n0), 0, 1);
    }
    v16bf a0 = frag_a(As[cur], wM, lane);
    v16bf a1 = frag_a(As[cur], wM + 16, lane);
    v16bf b0 = frag_b(Bt[cur], wN, lane);
    v16bf b1 = frag_b(Bt[cur], wN + 16, lane);
    acc[0][0] = wmma_bf16(a0, b0, acc[0][0]);
    acc[0][1] = wmma_bf16(a0, b1, acc[0][1]);
    acc[1][0] = wmma_bf16(a1, b0, acc[1][0]);
    acc[1][1] = wmma_bf16(a1, b1, acc[1][1]);
    if (more) {  // commit staged registers to the other buffer
      const int nxt = cur ^ 1;
#pragma unroll
      for (int it = 0; it < 2; ++it)
        *(v4us*)(&As[nxt][aR[it] * 40 + aK[it]]) = aReg[it];
#pragma unroll
      for (int it = 0; it < 4; ++it)
#pragma unroll
        for (int i = 0; i < 4; ++i)
          Bt[nxt][(bN[it] + i) * 40 + bK[it]] = bReg[it][i];
    }
    __syncthreads();
  }

#pragma unroll
  for (int i = 0; i < 2; ++i) {
#pragma unroll
    for (int r = 0; r < 8; ++r) {
      int m = m0 + wM + i * 16 + r + hi * 8;
      bool zrow = MASK ? (tokens[m] == 0) : false;
#pragma unroll
      for (int j = 0; j < 2; ++j) {
        int n = n0 + wN + j * 16 + l15;
        float val = acc[i][j][r];
        if (MASK && zrow) val = (n == 0) ? 1.0f : 0.0f;
        Out[(size_t)m * N + n] = val;
      }
    }
  }
}

// ---------------------------------------------------------------------------
// One recurrent timestep (fused): z = xproj[:,t,:] + h_prev @ Wh ; cell update.
// grid = 32 blocks (each owns 32 d-columns x 4 gates = 128 N cols), 256 thr.
// Double-buffered LDS; h double-buffered across step launches.
// ---------------------------------------------------------------------------
__global__ __launch_bounds__(256) void lstm_step_kernel(
    const float* __restrict__ xproj, const unsigned short* __restrict__ Whb,
    const float* __restrict__ h_prev, float* __restrict__ h_next,
    float* __restrict__ c_state, float* __restrict__ x_out,
    const int* __restrict__ tokens, int t, int is_first) {
  __shared__ __align__(16) unsigned short As[2][16 * 40];
  __shared__ __align__(16) unsigned short Bt[2][128 * 40];
  __shared__ float zl[4][16][33];
  const int tid = threadIdx.x, lane = tid & 31, wid = tid >> 5;
  const int g    = wid >> 1;            // gate 0..3 (i,f,g,o)
  const int dsub = (wid & 1) * 16;      // 16-col half of the 32-col d-slice
  const int hi   = (lane >> 4) & 1;
  const int l15  = lane & 15;
  const int d0   = blockIdx.x * 32;
  const int nG   = g * D_ + d0 + dsub + l15;  // column in 4D gate space

  v8f acc;
#pragma unroll
  for (int r = 0; r < 8; ++r) {
    int b = r + hi * 8;
    acc[r] = xproj[((size_t)b * T_ + t) * N4_ + nG];
  }

  if (!is_first) {
    // per-thread staging coordinates
    const int aR = (tid * 2) >> 5, aK = (tid * 2) & 31;  // 16 x 32 A tile
    int bK[4], bNl[4];
#pragma unroll
    for (int it = 0; it < 4; ++it) {
      int idx = (tid + it * 256) * 4;
      bK[it] = idx >> 7; bNl[it] = idx & 127;            // 32 x 128 B tile
    }
    v2f aReg;
    v4us bReg[4];
    auto bcol = [&](int nl) { return (nl >> 5) * D_ + d0 + (nl & 31); };

    // prologue: tile 0
    aReg = *(const v2f*)(h_prev + (size_t)aR * D_ + aK);
#pragma unroll
    for (int it = 0; it < 4; ++it)
      bReg[it] = *(const v4us*)(Whb + (size_t)bK[it] * N4_ + bcol(bNl[it]));
    As[0][aR * 40 + aK]     = f2bf(aReg[0]);
    As[0][aR * 40 + aK + 1] = f2bf(aReg[1]);
#pragma unroll
    for (int it = 0; it < 4; ++it)
#pragma unroll
      for (int i = 0; i < 4; ++i)
        Bt[0][(bNl[it] + i) * 40 + bK[it]] = bReg[it][i];
    __syncthreads();

    for (int k0 = 0; k0 < D_; k0 += 32) {
      const int cur = (k0 >> 5) & 1;
      const bool more = (k0 + 32) < D_;
      if (more) {
        int kk = k0 + 32;
        aReg = *(const v2f*)(h_prev + (size_t)aR * D_ + kk + aK);
#pragma unroll
        for (int it = 0; it < 4; ++it)
          bReg[it] =
              *(const v4us*)(Whb + (size_t)(kk + bK[it]) * N4_ + bcol(bNl[it]));
        if (kk + 32 < D_)
          __builtin_prefetch((const void*)(Whb + (size_t)(kk + 32) * N4_ + d0),
                             0, 1);
      }
      v16bf a = frag_a(As[cur], 0, lane);
      v16bf b = frag_b(Bt[cur], g * 32 + dsub, lane);
      acc = wmma_bf16(a, b, acc);
      if (more) {
        const int nxt = cur ^ 1;
        As[nxt][aR * 40 + aK]     = f2bf(aReg[0]);
        As[nxt][aR * 40 + aK + 1] = f2bf(aReg[1]);
#pragma unroll
        for (int it = 0; it < 4; ++it)
#pragma unroll
          for (int i = 0; i < 4; ++i)
            Bt[nxt][(bNl[it] + i) * 40 + bK[it]] = bReg[it][i];
      }
      __syncthreads();
    }
  }

  // scatter z tiles to LDS so all four gates of a d-column meet
#pragma unroll
  for (int r = 0; r < 8; ++r) zl[g][r + hi * 8][dsub + l15] = acc[r];
  __syncthreads();

  // elementwise cell update: 16 batches x 32 d = 512 elems, 2 per thread
#pragma unroll
  for (int e = 0; e < 2; ++e) {
    int idx = tid + e * 256;
    int b = idx >> 5, dl = idx & 31;
    int dg = d0 + dl;
    float i_ = sig_(zl[0][b][dl]);
    float f_ = sig_(zl[1][b][dl]);
    float g_ = tanh_(zl[2][b][dl]);
    float o_ = sig_(zl[3][b][dl]);
    float c_old = is_first ? 0.0f : c_state[(size_t)b * D_ + dg];
    float h_old = is_first ? 0.0f : h_prev[(size_t)b * D_ + dg];
    float c_new = f_ * c_old + i_ * g_;
    float h_new = o_ * tanh_(c_new);
    bool  msk   = tokens[b * T_ + t] != 0;
    float hO = msk ? h_new : h_old;
    float cO = msk ? c_new : c_old;
    c_state[(size_t)b * D_ + dg] = cO;
    h_next[(size_t)b * D_ + dg]  = hO;
    x_out[((size_t)b * T_ + t) * D_ + dg] = hO;
  }
}

// ---------------------------------------------------------------------------
extern "C" void kernel_launch(void* const* d_in, const int* in_sizes, int n_in,
                              void* d_out, int out_size, void* d_ws, size_t ws_size,
                              hipStream_t stream) {
  const int*   tokens = (const int*)d_in[0];
  const float* E      = (const float*)d_in[1];
  const float* Wx     = (const float*)d_in[2];
  const float* Wh     = (const float*)d_in[3];
  const float* bvec   = (const float*)d_in[4];
  const float* W_out  = (const float*)d_in[5];
  const float* b_out  = (const float*)d_in[6];
  float* out = (float*)d_out;

  const size_t nWx = (size_t)L_ * D_ * N4_;  // 8,388,608
  const size_t nWo = (size_t)D_ * V_;        // 32,768,000
  const size_t nX  = (size_t)M_ * D_;        // 4,194,304

  char* ws = (char*)d_ws;
  size_t off = 0;
  auto alloc = [&](size_t bytes) -> char* {
    char* p = ws + off;
    off += (bytes + 255) & ~(size_t)255;
    return p;
  };
  unsigned short* Wx_b = (unsigned short*)alloc(nWx * 2);
  unsigned short* Wh_b = (unsigned short*)alloc(nWx * 2);
  unsigned short* Wo_b = (unsigned short*)alloc(nWo * 2);
  unsigned short* Xb   = (unsigned short*)alloc(nX * 2);
  float* X    = (float*)alloc(nX * 4);
  float* XP   = (float*)alloc((size_t)M_ * N4_ * 4);
  float* Hbuf = (float*)alloc((size_t)2 * B_ * D_ * 4);
  float* Cbuf = (float*)alloc((size_t)B_ * D_ * 4);

  // 1) weight conversion f32 -> bf16
  {
    int n4 = (int)(nWx / 4);
    cvt_bf16_kernel<<<(n4 + 255) / 256, 256, 0, stream>>>(Wx, Wx_b, n4);
    cvt_bf16_kernel<<<(n4 + 255) / 256, 256, 0, stream>>>(Wh, Wh_b, n4);
    int n4o = (int)(nWo / 4);
    cvt_bf16_kernel<<<(n4o + 255) / 256, 256, 0, stream>>>(W_out, Wo_b, n4o);
  }

  // 2) embedding gather
  embed_kernel<<<M_, 256, 0, stream>>>(tokens, E, X);

  const int n4x = (int)(nX / 4);

  // 3) LSTM layers
  for (int l = 0; l < L_; ++l) {
    const unsigned short* Wxl = Wx_b + (size_t)l * D_ * N4_;
    const unsigned short* Whl = Wh_b + (size_t)l * D_ * N4_;
    const float* bl = bvec + (size_t)l * N4_;

    // 3a) layer input -> bf16, then xproj = Xb @ Wx + b (whole sequence)
    cvt_bf16_kernel<<<(n4x + 255) / 256, 256, 0, stream>>>(X, Xb, n4x);
    big_gemm_kernel<N4_, false><<<dim3(N4_ / 128, M_ / 64), 256, 0, stream>>>(
        Xb, Wxl, bl, nullptr, XP);

    // 3b) 256 sequential recurrent steps (stream-ordered, h double-buffered)
    for (int t = 0; t < T_; ++t) {
      int s = l * T_ + t;
      const float* hp = Hbuf + (size_t)(s & 1) * B_ * D_;
      float*       hn = Hbuf + (size_t)((s + 1) & 1) * B_ * D_;
      lstm_step_kernel<<<D_ / 32, 256, 0, stream>>>(XP, Whl, hp, hn, Cbuf, X,
                                                    tokens, t, t == 0 ? 1 : 0);
    }
  }

  // 4) output projection + masking
  cvt_bf16_kernel<<<(n4x + 255) / 256, 256, 0, stream>>>(X, Xb, n4x);
  big_gemm_kernel<V_, true><<<dim3(V_ / 128, M_ / 64), 256, 0, stream>>>(
      Xb, Wo_b, b_out, tokens, out);
}
